// SelfAttentionModule_17008070492459
// MI455X (gfx1250) — compile-verified, matrix operand order checked by reading, not run on hardware
//
#include <hip/hip_runtime.h>

typedef __attribute__((ext_vector_type(16))) __bf16 v16bf;
typedef __attribute__((ext_vector_type(8)))  __bf16 v8bf;
typedef __attribute__((ext_vector_type(4)))  __bf16 v4bf;
typedef __attribute__((ext_vector_type(2)))  __bf16 v2bf;
typedef __attribute__((ext_vector_type(8)))  float  v8f;

#define WMMA_BF16(a, b, c) \
  __builtin_amdgcn_wmma_f32_16x16x32_bf16(false, (a), false, (b), (short)0, (c), false, false)

#if __has_builtin(__builtin_amdgcn_global_load_async_to_lds_b128) && \
    __has_builtin(__builtin_amdgcn_s_wait_asynccnt)
#define HAVE_ASYNC_LDS 1
// Exact parameter types per hipcc diagnostic: v4i in AS1 / AS3, imm, imm.
typedef int v4i_vs __attribute__((vector_size(16)));
typedef __attribute__((address_space(1))) v4i_vs* gbl_v4i_p;
typedef __attribute__((address_space(3))) v4i_vs* lds_v4i_p;
#else
#define HAVE_ASYNC_LDS 0
#endif

constexpr int Bv = 4;     // batch
constexpr int Cv = 256;   // channels
constexpr int Nv = 4096;  // H*W
constexpr int Dv = 32;    // q/k dim

// ---------------------------------------------------------------------------
// 1) fp32 -> bf16 convert of x, vectorized: b128 load, packed cvt, b64 store
// ---------------------------------------------------------------------------
__global__ void k_convert(const float4* __restrict__ x, v4bf* __restrict__ xb, int n4) {
  int i = blockIdx.x * blockDim.x + threadIdx.x;
  if (i < n4) {
    const float4 v = x[i];
    v4bf o;
    o[0] = (__bf16)v.x; o[1] = (__bf16)v.y; o[2] = (__bf16)v.z; o[3] = (__bf16)v.w;
    xb[i] = o;
  }
}

// ---------------------------------------------------------------------------
// 2) Projection GEMM: Y[d,n] = sum_c W[d,c] * X[c,n] + bias[d]   (per batch)
//    One wave per 16x16 output tile, one v_wmma_f32_16x16x32_bf16 per 32-wide
//    K step over C. Templated so the store path is branch-free; the TRANS
//    (q/k) path stores 8 consecutive bf16 along D as a single b128.
// ---------------------------------------------------------------------------
template <int DOUT, bool TRANS>
__global__ void k_proj(const __bf16* __restrict__ xb, const float* __restrict__ W,
                       const float* __restrict__ bias, __bf16* __restrict__ out) {
  __shared__ __attribute__((aligned(32))) __bf16 lds[32 * 16];  // [col][row]
  const int lane = threadIdx.x;
  const int n0 = blockIdx.x * 16, d0 = blockIdx.y * 16, b = blockIdx.z;

  v8f acc = {};
  for (int c0 = 0; c0 < Cv; c0 += 32) {
    // A fragment: weight rows, fp32 -> bf16 in flight
    v16bf a;
    const float* wrow = W + (size_t)(d0 + (lane & 15)) * Cv + c0 + 16 * (lane >> 4);
#pragma unroll
    for (int t = 0; t < 16; ++t) a[t] = (__bf16)wrow[t];

    // Stage X tile (32 c-rows x 16 n-cols) transposed into LDS
    const __bf16* xrow = xb + ((size_t)b * Cv + c0 + lane) * Nv + n0;
#pragma unroll
    for (int t = 0; t < 16; ++t) lds[t * 32 + lane] = xrow[t];
    __syncthreads();

    const v16bf bm = *(const v16bf*)&lds[(lane & 15) * 32 + 16 * (lane >> 4)];
    acc = WMMA_BF16(a, bm, acc);
    __syncthreads();
  }

  const int col = n0 + (lane & 15);
  if constexpr (TRANS) {
    // rows d0+8*(lane>>4)+r, r=0..7 are consecutive in [N, DOUT] layout
    v8bf y;
#pragma unroll
    for (int r = 0; r < 8; ++r)
      y[r] = (__bf16)(acc[r] + bias[d0 + 8 * (lane >> 4) + r]);
    *(v8bf*)(out + ((size_t)b * Nv + col) * DOUT + d0 + 8 * (lane >> 4)) = y;
  } else {
#pragma unroll
    for (int r = 0; r < 8; ++r) {
      const int row = d0 + r + 8 * (lane >> 4);
      out[((size_t)b * DOUT + row) * Nv + col] = (__bf16)(acc[r] + bias[row]);
    }
  }
}

// ---------------------------------------------------------------------------
// 3) Row softmax stats: m[i] = max_j q_i.k_j ; s[i] = sum_j exp(e - m).
//    Block = 16 query rows; 8 waves stream 256 key tiles (one WMMA each,
//    D=32 == one K step). Online max/sum per lane, shfl_xor merge across the
//    16 lanes sharing a row, then LDS merge across waves.
// ---------------------------------------------------------------------------
__global__ void k_stats(const __bf16* __restrict__ qT, const __bf16* __restrict__ kT,
                        float* __restrict__ mrow, float* __restrict__ srow) {
  __shared__ float pm[8 * 16], ps[8 * 16];
  const int lane = threadIdx.x & 31, wave = threadIdx.x >> 5;
  const int i0 = blockIdx.x * 16, b = blockIdx.y;

  const __bf16* qp = qT + ((size_t)b * Nv + i0 + (lane & 15)) * Dv + 16 * (lane >> 4);
  const v16bf a = *(const v16bf*)qp;  // A tile fixed for whole block

  float m[8], s[8];
#pragma unroll
  for (int r = 0; r < 8; ++r) { m[r] = -__builtin_inff(); s[r] = 0.f; }

  for (int t = 0; t < 32; ++t) {
    const int j0 = (t * 8 + wave) * 16;
    const v16bf kb =
        *(const v16bf*)(kT + ((size_t)b * Nv + j0 + (lane & 15)) * Dv + 16 * (lane >> 4));
    v8f z = {};
    v8f e = WMMA_BF16(a, kb, z);
#pragma unroll
    for (int r = 0; r < 8; ++r) {
      const float v = e[r];
      const float mn = fmaxf(m[r], v);
      s[r] = s[r] * __expf(m[r] - mn) + __expf(v - mn);
      m[r] = mn;
    }
  }

#pragma unroll
  for (int mask = 1; mask <= 8; mask <<= 1) {
#pragma unroll
    for (int r = 0; r < 8; ++r) {
      const float mo = __shfl_xor(m[r], mask, 32);
      const float so = __shfl_xor(s[r], mask, 32);
      const float mn = fmaxf(m[r], mo);
      s[r] = s[r] * __expf(m[r] - mn) + so * __expf(mo - mn);
      m[r] = mn;
    }
  }
  if ((lane & 15) == 0) {
#pragma unroll
    for (int r = 0; r < 8; ++r) {
      pm[wave * 16 + 8 * (lane >> 4) + r] = m[r];
      ps[wave * 16 + 8 * (lane >> 4) + r] = s[r];
    }
  }
  __syncthreads();
  if (threadIdx.x < 16) {
    float mm = -__builtin_inff(), ss = 0.f;
#pragma unroll
    for (int w = 0; w < 8; ++w) {
      const float mo = pm[w * 16 + threadIdx.x], so = ps[w * 16 + threadIdx.x];
      const float mn = fmaxf(mm, mo);
      ss = ss * __expf(mm - mn) + so * __expf(mo - mn);
      mm = mn;
    }
    mrow[(size_t)b * Nv + i0 + threadIdx.x] = mm;
    srow[(size_t)b * Nv + i0 + threadIdx.x] = ss;
  }
}

// ---------------------------------------------------------------------------
// 4) Fused output GEMM: out[c,j] = gamma * sum_i softmax(e)[i,j] * v[c,i] + x.
//    Block = (b, 32-col j tile), 256 threads = 8 waves covering all 256 c.
//    Stats (m, 1/s) are staged into LDS with async global->LDS b128 copies
//    (ASYNCcnt). Per 32-i step: waves 0..3 recompute the 32x32 energy block
//    (1 WMMA each), apply exp * 1/s, deposit a bf16 P tile in LDS (packed b32
//    stores); then every wave issues 4 accumulation WMMAs, reusing each vb
//    fragment against two P fragments.
// ---------------------------------------------------------------------------
__global__ void k_out(const __bf16* __restrict__ qT, const __bf16* __restrict__ kT,
                      const __bf16* __restrict__ vb, const float* __restrict__ mrow,
                      const float* __restrict__ srow, const float* __restrict__ x,
                      const float* __restrict__ gamma, float* __restrict__ outp) {
  __shared__ float sm[Nv];
  __shared__ float si[Nv];
  __shared__ __attribute__((aligned(32))) __bf16 pt[32 * 32];  // [j local][i local]
  const int lane = threadIdx.x & 31, wave = threadIdx.x >> 5;
  const int j0 = blockIdx.x * 32, b = blockIdx.y;
  const int jsub = wave & 1, isub = wave >> 1;  // energy-tile assignment, waves 0..3

#if HAVE_ASYNC_LDS
  // Bulk-stage raw stats via the async DMA path (tracked by ASYNCcnt), then
  // invert s in place.
  for (int i4 = threadIdx.x; i4 < Nv / 4; i4 += 256) {
    __builtin_amdgcn_global_load_async_to_lds_b128(
        (gbl_v4i_p)(mrow + (size_t)b * Nv + i4 * 4), (lds_v4i_p)&sm[i4 * 4], 0, 0);
    __builtin_amdgcn_global_load_async_to_lds_b128(
        (gbl_v4i_p)(srow + (size_t)b * Nv + i4 * 4), (lds_v4i_p)&si[i4 * 4], 0, 0);
  }
  __builtin_amdgcn_s_wait_asynccnt(0);
  __syncthreads();
  for (int idx = threadIdx.x; idx < Nv; idx += 256) si[idx] = 1.0f / si[idx];
#else
  for (int idx = threadIdx.x; idx < Nv; idx += 256) {
    sm[idx] = mrow[(size_t)b * Nv + idx];
    si[idx] = 1.0f / srow[(size_t)b * Nv + idx];
  }
#endif
  // Loop-invariant energy B fragment (k columns for this wave's j sub-tile)
  v16bf kb = {};
  if (wave < 4)
    kb = *(const v16bf*)(kT + ((size_t)b * Nv + j0 + jsub * 16 + (lane & 15)) * Dv +
                         16 * (lane >> 4));
  v8f acc[2][2] = {};  // [c sub-tile][j sub-tile]
  __syncthreads();

  for (int i0 = 0; i0 < Nv; i0 += 32) {
    if (wave < 4) {
      const __bf16* qp = qT + ((size_t)b * Nv + i0 + isub * 16 + (lane & 15)) * Dv +
                         16 * (lane >> 4);
      const v16bf a = *(const v16bf*)qp;
      v8f z = {};
      v8f e = WMMA_BF16(a, kb, z);
      const int jl = jsub * 16 + (lane & 15);
#pragma unroll
      for (int r = 0; r < 8; r += 2) {
        const int il = isub * 16 + r + 8 * (lane >> 4);
        const int i = i0 + il;
        v2bf pr;
        pr[0] = (__bf16)(__expf(e[r] - sm[i]) * si[i]);
        pr[1] = (__bf16)(__expf(e[r + 1] - sm[i + 1]) * si[i + 1]);
        *(v2bf*)&pt[jl * 32 + il] = pr;  // il even -> 4B-aligned packed store
      }
    }
    __syncthreads();
    {
      const __bf16* vp0 =
          vb + ((size_t)b * Cv + wave * 32 + (lane & 15)) * Nv + i0 + 16 * (lane >> 4);
      const v16bf pa = *(const v16bf*)&pt[(lane & 15) * 32 + 16 * (lane >> 4)];
      const v16bf pb = *(const v16bf*)&pt[(16 + (lane & 15)) * 32 + 16 * (lane >> 4)];
      const v16bf a0 = *(const v16bf*)vp0;
      acc[0][0] = WMMA_BF16(a0, pa, acc[0][0]);
      acc[0][1] = WMMA_BF16(a0, pb, acc[0][1]);
      const v16bf a1 = *(const v16bf*)(vp0 + (size_t)16 * Nv);
      acc[1][0] = WMMA_BF16(a1, pa, acc[1][0]);
      acc[1][1] = WMMA_BF16(a1, pb, acc[1][1]);
      __builtin_prefetch(vp0 + 32, 0, 0);  // next i-step -> global_prefetch_b8
    }
    __syncthreads();
  }

  const float g = gamma[0];
#pragma unroll
  for (int ct = 0; ct < 2; ++ct) {
#pragma unroll
    for (int jt = 0; jt < 2; ++jt) {
      const int col = j0 + jt * 16 + (lane & 15);
#pragma unroll
      for (int r = 0; r < 8; ++r) {
        const int c = wave * 32 + ct * 16 + r + 8 * (lane >> 4);
        const size_t idx = ((size_t)b * Cv + c) * Nv + col;
        outp[idx] = g * acc[ct][jt][r] + x[idx];
      }
    }
  }
}

// ---------------------------------------------------------------------------
extern "C" void kernel_launch(void* const* d_in, const int* in_sizes, int n_in,
                              void* d_out, int out_size, void* d_ws, size_t ws_size,
                              hipStream_t stream) {
  (void)in_sizes; (void)n_in; (void)out_size; (void)ws_size;
  const float* x  = (const float*)d_in[0];
  const float* wq = (const float*)d_in[1];
  const float* bq = (const float*)d_in[2];
  const float* wk = (const float*)d_in[3];
  const float* bk = (const float*)d_in[4];
  const float* wv = (const float*)d_in[5];
  const float* bv = (const float*)d_in[6];
  const float* gm = (const float*)d_in[7];

  char* ws = (char*)d_ws;
  size_t off = 0;
  __bf16* xb = (__bf16*)(ws + off); off += (size_t)Bv * Cv * Nv * 2;  // 8 MB
  __bf16* vb = (__bf16*)(ws + off); off += (size_t)Bv * Cv * Nv * 2;  // 8 MB
  __bf16* qT = (__bf16*)(ws + off); off += (size_t)Bv * Nv * Dv * 2;  // 1 MB
  __bf16* kT = (__bf16*)(ws + off); off += (size_t)Bv * Nv * Dv * 2;  // 1 MB
  float* mrow = (float*)(ws + off); off += (size_t)Bv * Nv * 4;
  float* srow = (float*)(ws + off); off += (size_t)Bv * Nv * 4;

  const int total4 = (Bv * Cv * Nv) / 4;
  k_convert<<<(total4 + 255) / 256, 256, 0, stream>>>((const float4*)x, (v4bf*)xb, total4);
  k_proj<Dv, true><<<dim3(Nv / 16, Dv / 16, Bv), 32, 0, stream>>>(xb, wq, bq, qT);
  k_proj<Dv, true><<<dim3(Nv / 16, Dv / 16, Bv), 32, 0, stream>>>(xb, wk, bk, kT);
  k_proj<Cv, false><<<dim3(Nv / 16, Cv / 16, Bv), 32, 0, stream>>>(xb, wv, bv, vb);
  k_stats<<<dim3(Nv / 16, Bv), 256, 0, stream>>>(qT, kT, mrow, srow);
  k_out<<<dim3(Nv / 32, Bv), 256, 0, stream>>>(qT, kT, vb, mrow, srow, x, gm, (float*)d_out);
}
